// EGNNLayer_29566554866417
// MI455X (gfx1250) — matrix operand order repaired and checked by reading, not compiled, not run on hardware
//
#include <hip/hip_runtime.h>

typedef __attribute__((ext_vector_type(16))) _Float16 v16h;
typedef __attribute__((ext_vector_type(8)))  _Float16 v8h;
typedef __attribute__((ext_vector_type(8)))  float    v8f;

#define NN 512

// Compiler memory barrier: blocks LICM from hoisting (and spilling) the
// L2-resident weight-fragment loads out of the j-tile loop.
#define KBAR() asm volatile("" ::: "memory")

// ---- packed-weight offsets (in halves) inside d_ws --------------------------
constexpr size_t OFF_MSG_W1   = 0;        // K=192(161) N=128 : 6x8 tiles
constexpr size_t OFF_MSG_W2   = 24576;    // 4x8
constexpr size_t OFF_MSG_W3   = 40960;    // 4x8
constexpr size_t OFF_ATTN_W1  = 57344;    // 4x4
constexpr size_t OFF_VAL_W    = 65536;    // 4x8
constexpr size_t OFF_COORD_W1 = 81920;    // 4x4
constexpr size_t OFF_EDGE_W1  = 90112;    // 5x8 (K=160)
constexpr size_t OFF_EDGE_W2  = 110592;   // 4x2

// ---- LDS layout -------------------------------------------------------------
constexpr int W_STAGE_A  = 0;       // 16x192 f16 (6144 B)
constexpr int W_STAGE_B  = 6144;    // 16x160 f16 (5120 B)
constexpr int W_ACT      = 11264;   // 16x128 f32 (8192 B)
constexpr int W_LOGITS   = 19456;   // 16x4 f32
constexpr int W_PEXP     = 19712;   // 16x4 f32
constexpr int W_XDN      = 19968;   // 16x2 f32
constexpr int WAVE_BYTES = 20096;
constexpr int B_COMBACC  = WAVE_BYTES * 4;     // 4x128 f32
constexpr int B_COMBM    = B_COMBACC + 2048;   // 4x4
constexpr int B_COMBL    = B_COMBM + 64;       // 4x4
constexpr int B_COMBCW   = B_COMBL + 64;       // 4x2
constexpr int B_GM       = B_COMBCW + 32;      // 4
constexpr int B_GL       = B_GM + 16;          // 4
constexpr int B_HIN      = B_GL + 16;          // 192 f32
constexpr int B_TBUF     = B_HIN + 768;        // 128 f32
constexpr int B_STATS    = B_TBUF + 512;       // 4 f32
constexpr int B_HRES     = B_STATS + 16;       // 64 f32
constexpr int SMEM_BYTES = B_HRES + 256;       // 84176 B

__device__ __forceinline__ float fast_rcp(float v)  { return __builtin_amdgcn_rcpf(v); }
__device__ __forceinline__ float silu(float v)      { return v * fast_rcp(1.0f + __expf(-v)); }
__device__ __forceinline__ float fast_tanh(float z) {
  z = fminf(fmaxf(z, -10.0f), 10.0f);
  float ex = __expf(2.0f * z);
  return (ex - 1.0f) * fast_rcp(ex + 1.0f);
}

// convert 8 consecutive f32 (16B-aligned) to 8 f16 and store as one b128
__device__ __forceinline__ void cvt8(_Float16* dst, const float* src) {
  float4 a = ((const float4*)src)[0];
  float4 b = ((const float4*)src)[1];
  v8h o;
  o[0] = (_Float16)a.x; o[1] = (_Float16)a.y; o[2] = (_Float16)a.z; o[3] = (_Float16)a.w;
  o[4] = (_Float16)b.x; o[5] = (_Float16)b.y; o[6] = (_Float16)b.z; o[7] = (_Float16)b.w;
  *(v8h*)dst = o;
}

// ---- weight repack: f32 (K,N) -> f16 WMMA-B fragment layout -----------------
__global__ void pack_weights(const float* __restrict__ W, _Float16* __restrict__ dst,
                             int K, int Kpad, int N) {
  int idx = blockIdx.x * blockDim.x + threadIdx.x;
  int total = (Kpad >> 5) * (N >> 4) * 512;
  if (idx >= total) return;
  int e    = idx & 15;
  int lane = (idx >> 4) & 31;
  int tile = idx >> 9;
  int ntn  = N >> 4;
  int kt   = tile / ntn, nt = tile - kt * ntn;
  int n    = nt * 16 + (lane & 15);
  int kl   = ((lane >> 4) << 3) + (e & 7) + ((e >> 3) << 4);  // B layout: lanes 0-15 K=0..7,16..23; lanes 16-31 K=8..15,24..31
  int k    = (kt << 5) + kl;
  float v  = (k < K) ? W[(size_t)k * N + n] : 0.0f;
  dst[idx] = (_Float16)v;
}

// ---- A fragment from LDS row-major staging ----------------------------------
__device__ __forceinline__ v16h a_frag(const _Float16* base, int stride, int lane, int kbase) {
  const _Float16* p = base + (lane & 15) * stride + kbase + ((lane >> 4) << 3);
  v8h lo = *(const v8h*)p;
  v8h hi = *(const v8h*)(p + 16);
  v16h r;
#pragma unroll
  for (int q = 0; q < 8; ++q) { r[q] = lo[q]; r[8 + q] = hi[q]; }
  return r;
}

template <int NTN, int KTN, bool SILU>
__device__ __forceinline__ void gemm_f32o(const _Float16* A, int astride,
                                          const _Float16* __restrict__ W,
                                          const float* __restrict__ bias,
                                          float* out, int ostride, int lane) {
  const int m0 = (lane >> 4) << 3;
  const int nl = lane & 15;
#pragma unroll
  for (int nt = 0; nt < NTN; ++nt) {
    float bv = bias[nt * 16 + nl];
    v8f c;
#pragma unroll
    for (int r = 0; r < 8; ++r) c[r] = bv;
#pragma unroll
    for (int kt = 0; kt < KTN; ++kt) {
      v16h a  = a_frag(A, astride, lane, kt * 32);
      v16h bw = *(const v16h*)(W + (size_t)(kt * NTN + nt) * 512 + lane * 16);
      c = __builtin_amdgcn_wmma_f32_16x16x32_f16(false, a, false, bw, (short)0, c, false, false);
    }
#pragma unroll
    for (int r = 0; r < 8; ++r) {
      float v = c[r];
      if (SILU) v = silu(v);
      out[(m0 + r) * ostride + nt * 16 + nl] = v;
    }
  }
}

template <int NTN, int KTN, bool SILU>
__device__ __forceinline__ void gemm_f16o(const _Float16* A, int astride,
                                          const _Float16* __restrict__ W,
                                          const float* __restrict__ bias,
                                          _Float16* out, int ostride, int lane) {
  const int m0 = (lane >> 4) << 3;
  const int nl = lane & 15;
#pragma unroll
  for (int nt = 0; nt < NTN; ++nt) {
    float bv = bias[nt * 16 + nl];
    v8f c;
#pragma unroll
    for (int r = 0; r < 8; ++r) c[r] = bv;
#pragma unroll
    for (int kt = 0; kt < KTN; ++kt) {
      v16h a  = a_frag(A, astride, lane, kt * 32);
      v16h bw = *(const v16h*)(W + (size_t)(kt * NTN + nt) * 512 + lane * 16);
      c = __builtin_amdgcn_wmma_f32_16x16x32_f16(false, a, false, bw, (short)0, c, false, false);
    }
#pragma unroll
    for (int r = 0; r < 8; ++r) {
      float v = c[r];
      if (SILU) v = silu(v);
      out[(m0 + r) * ostride + nt * 16 + nl] = (_Float16)v;
    }
  }
}

// LN over 128 features, 2 lanes per row (shfl-combined), writes f16
__device__ __forceinline__ void ln128_to_f16(const float* in, int istride,
                                             const float* __restrict__ g,
                                             const float* __restrict__ bta,
                                             _Float16* out, int ostride, int lane) {
  const int row  = lane & 15;
  const int coff = (lane >> 4) * 64;
  const float* rp = in + row * istride + coff;
  float s = 0.f;
  for (int c = 0; c < 64; ++c) s += rp[c];
  s += __shfl_xor(s, 16);
  float m = s * (1.0f / 128.0f);
  float v = 0.f;
  for (int c = 0; c < 64; ++c) { float d = rp[c] - m; v += d * d; }
  v += __shfl_xor(v, 16);
  float inv = rsqrtf(v * (1.0f / 128.0f) + 1e-5f);
  _Float16* op = out + row * ostride + coff;
  const float* gg = g + coff;
  const float* bb = bta + coff;
  for (int c = 0; c < 64; ++c)
    op[c] = (_Float16)((rp[c] - m) * inv * gg[c] + bb[c]);
}

// ---- fused per-(b,i) kernel -------------------------------------------------
__global__ __launch_bounds__(128, 1) void egnn_main(
    const float* __restrict__ h, const float* __restrict__ x, const float* __restrict__ e,
    const _Float16* __restrict__ wp,
    const float* __restrict__ msg_b1, const float* __restrict__ msg_ln_g, const float* __restrict__ msg_ln_b,
    const float* __restrict__ msg_b2, const float* __restrict__ msg_b3,
    const float* __restrict__ attn_b1, const float* __restrict__ attn_W2, const float* __restrict__ attn_b2,
    const float* __restrict__ val_b,
    const float* __restrict__ coord_b1, const float* __restrict__ coord_W2,
    const float* __restrict__ edge_b1, const float* __restrict__ edge_ln_g, const float* __restrict__ edge_ln_b,
    const float* __restrict__ edge_b2, const float* __restrict__ en_g, const float* __restrict__ en_b,
    const float* __restrict__ node_W1, const float* __restrict__ node_b1,
    const float* __restrict__ node_ln_g, const float* __restrict__ node_ln_b,
    const float* __restrict__ node_W2, const float* __restrict__ node_b2,
    const float* __restrict__ nn_g, const float* __restrict__ nn_b,
    const float* __restrict__ coord_scale,
    float* __restrict__ out_h, float* __restrict__ out_x, float* __restrict__ out_e) {
  extern __shared__ char smem[];
  const int tid  = threadIdx.x;
  const int lane = tid & 31;
  const int wave = tid >> 5;
  const int bi   = blockIdx.x;
  const int b    = bi >> 9;
  const int i    = bi & (NN - 1);

  char*     wbase  = smem + wave * WAVE_BYTES;
  _Float16* stA    = (_Float16*)(wbase + W_STAGE_A);   // 16x192
  _Float16* stB    = (_Float16*)(wbase + W_STAGE_B);   // 16x160
  float*    act    = (float*)(wbase + W_ACT);          // 16x128
  float*    logits = (float*)(wbase + W_LOGITS);
  float*    pexp   = (float*)(wbase + W_PEXP);
  float*    xdn    = (float*)(wbase + W_XDN);

  float* combacc = (float*)(smem + B_COMBACC);
  float* combm   = (float*)(smem + B_COMBM);
  float* combl   = (float*)(smem + B_COMBL);
  float* combcw  = (float*)(smem + B_COMBCW);
  float* gm      = (float*)(smem + B_GM);
  float* gl      = (float*)(smem + B_GL);
  float* hin     = (float*)(smem + B_HIN);
  float* tbuf    = (float*)(smem + B_TBUF);
  float* stats   = (float*)(smem + B_STATS);
  float* hres    = (float*)(smem + B_HRES);

  const float* hi_row = h + ((size_t)b * NN + i) * 64;
  const float  xi0    = x[((size_t)b * NN + i) * 2 + 0];
  const float  xi1    = x[((size_t)b * NN + i) * 2 + 1];

  float mh[4], lh[4], acc8[8];
#pragma unroll
  for (int k = 0; k < 4; ++k) { mh[k] = -3e38f; lh[k] = 0.f; }
#pragma unroll
  for (int k = 0; k < 8; ++k) acc8[k] = 0.f;
  float cw0 = 0.f, cw1 = 0.f;

  const int r1   = lane >> 1;   // staging: one row per lane-pair
  const int half = lane & 1;
  const int row  = lane & 15;   // parallel LN/dot helpers: one row per lane-pair (xor 16)
  const int hi16 = lane >> 4;

#pragma unroll 1
  for (int jt = wave; jt < 32; jt += 4) {
    KBAR();  // block cross-iteration hoisting of weight loads
    const int j0 = jt * 16;
    // ---- stage msg_input = [h_i | h_j | e | dist | 0-pad] as f16 (16x192) ---
    {
      int j = j0 + r1;
      const float* hj = h + ((size_t)b * NN + j) * 64;
      const float* er = e + (((size_t)b * NN + i) * NN + j) * 32;
      _Float16* rowp  = stA + r1 * 192;
#pragma unroll
      for (int c8 = 0; c8 < 4; ++c8) {   // h_i : cols 0..63 (32 per lane)
        int c = half * 32 + c8 * 8;
        cvt8(rowp + c, hi_row + c);
      }
#pragma unroll
      for (int c8 = 0; c8 < 4; ++c8) {   // h_j : cols 64..127
        int c = half * 32 + c8 * 8;
        cvt8(rowp + 64 + c, hj + c);
      }
#pragma unroll
      for (int c8 = 0; c8 < 2; ++c8) {   // e : cols 128..159 (16 per lane)
        int c = half * 16 + c8 * 8;
        cvt8(rowp + 128 + c, er + c);
      }
      v8h z = {};
      if (half == 0) {
        *(v8h*)(rowp + 160) = z;         // cols 160..175 zero
        *(v8h*)(rowp + 168) = z;
        float dx = x[((size_t)b * NN + j) * 2 + 0] - xi0;
        float dy = x[((size_t)b * NN + j) * 2 + 1] - xi1;
        float dist = sqrtf(dx * dx + dy * dy);
        rowp[160] = (_Float16)dist;      // overwrite col 160 with dist
        float inv = fast_rcp(dist + 1e-8f);
        xdn[r1 * 2 + 0] = dx * inv;
        xdn[r1 * 2 + 1] = dy * inv;
      } else {
        *(v8h*)(rowp + 176) = z;         // cols 176..191 zero
        *(v8h*)(rowp + 184) = z;
      }
    }
    KBAR();
    // ---- message MLP --------------------------------------------------------
    gemm_f32o<8, 6, true>(stA, 192, wp + OFF_MSG_W1, msg_b1, act, 128, lane);
    KBAR();
    ln128_to_f16(act, 128, msg_ln_g, msg_ln_b, stB, 160, lane);
    KBAR();
    gemm_f16o<8, 4, true>(stB, 160, wp + OFF_MSG_W2, msg_b2, stA, 192, lane);
    KBAR();
    gemm_f16o<8, 4, false>(stA, 192, wp + OFF_MSG_W3, msg_b3, stB + 32, 160, lane); // messages
    {  // e(f16) -> edge-input cols 0..31 (16 halves per lane)
      const _Float16* src = stA + row * 192 + 128 + hi16 * 16;
      _Float16*       dst = stB + row * 160 + hi16 * 16;
      *(v8h*)dst       = *(const v8h*)src;
      *(v8h*)(dst + 8) = *(const v8h*)(src + 8);
    }
    KBAR();
    // ---- attention logits ---------------------------------------------------
    gemm_f32o<4, 4, true>(stB + 32, 160, wp + OFF_ATTN_W1, attn_b1, act, 128, lane);
    KBAR();
    {  // 64 dots of length 64: 2 per lane (row, head-pair)
      int hp = hi16 * 2;
#pragma unroll
      for (int dh = 0; dh < 2; ++dh) {
        int hh = hp + dh;
        float s = attn_b2[hh];
        for (int k = 0; k < 64; ++k) s += act[row * 128 + k] * attn_W2[k * 4 + hh];
        logits[row * 4 + hh] = s * 0.17677669529663687f;  // 1/sqrt(32)
      }
    }
    KBAR();
    // ---- online softmax update ---------------------------------------------
    float scl[4];
#pragma unroll
    for (int hh = 0; hh < 4; ++hh) {
      float tm = -3e38f;
      for (int r = 0; r < 16; ++r) tm = fmaxf(tm, logits[r * 4 + hh]);
      float nm = fmaxf(mh[hh], tm);
      scl[hh]  = __expf(mh[hh] - nm);
      mh[hh]   = nm;
    }
    if (lane < 16) {
      for (int hh = 0; hh < 4; ++hh)
        pexp[lane * 4 + hh] = __expf(logits[lane * 4 + hh] - mh[hh]);
    }
    KBAR();
#pragma unroll
    for (int hh = 0; hh < 4; ++hh) {
      float rs = 0.f;
      for (int r = 0; r < 16; ++r) rs += pexp[r * 4 + hh];
      lh[hh] = lh[hh] * scl[hh] + rs;
    }
    // ---- values GEMM fused with weighted reduction over rows ---------------
    {
      const int m0 = hi16 << 3;
      const int nl = lane & 15;
#pragma unroll
      for (int nt = 0; nt < 8; ++nt) {
        int col = nt * 16 + nl;
        int hh  = col >> 5;
        v8f c;
        float bv = val_b[col];
#pragma unroll
        for (int r = 0; r < 8; ++r) c[r] = bv;
#pragma unroll
        for (int kt = 0; kt < 4; ++kt) {
          v16h a  = a_frag(stB + 32, 160, lane, kt * 32);
          v16h bw = *(const v16h*)(wp + OFF_VAL_W + (size_t)(kt * 8 + nt) * 512 + lane * 16);
          c = __builtin_amdgcn_wmma_f32_16x16x32_f16(false, a, false, bw, (short)0, c, false, false);
        }
        float s = acc8[nt] * scl[hh];
#pragma unroll
        for (int r = 0; r < 8; ++r) s += pexp[(m0 + r) * 4 + hh] * c[r];
        acc8[nt] = s;
      }
    }
    KBAR();
    // ---- coordinate weights -------------------------------------------------
    gemm_f32o<4, 4, true>(stB + 32, 160, wp + OFF_COORD_W1, coord_b1, act, 128, lane);
    KBAR();
    {  // half-dot per lane, combine across the lane pair
      int koff = hi16 * 32;
      float s = 0.f;
      for (int k = 0; k < 32; ++k) s += act[row * 128 + koff + k] * coord_W2[koff + k];
      s += __shfl_xor(s, 16);
      if (lane < 16) {
        float cwv = fast_tanh(s * 0.1f);
        cw0 += cwv * xdn[row * 2 + 0];
        cw1 += cwv * xdn[row * 2 + 1];
      }
    }
    KBAR();
    // ---- edge MLP + residual LN, stream out --------------------------------
    gemm_f32o<8, 5, true>(stB, 160, wp + OFF_EDGE_W1, edge_b1, act, 128, lane);
    KBAR();
    ln128_to_f16(act, 128, edge_ln_g, edge_ln_b, stA, 192, lane);
    KBAR();
    gemm_f32o<2, 4, false>(stA, 192, wp + OFF_EDGE_W2, edge_b2, act, 128, lane);
    KBAR();
    {  // residual + LN over 32, one row per lane pair, 16 cols per lane
      int j = j0 + row;
      int c0 = hi16 * 16;
      size_t eoff = (((size_t)b * NN + i) * NN + j) * 32 + c0;
      float t[16];
      float s1 = 0.f, s2 = 0.f;
#pragma unroll
      for (int q = 0; q < 4; ++q) {
        float4 ev = ((const float4*)(e + eoff))[q];
        t[q * 4 + 0] = ev.x + act[row * 128 + c0 + q * 4 + 0];
        t[q * 4 + 1] = ev.y + act[row * 128 + c0 + q * 4 + 1];
        t[q * 4 + 2] = ev.z + act[row * 128 + c0 + q * 4 + 2];
        t[q * 4 + 3] = ev.w + act[row * 128 + c0 + q * 4 + 3];
      }
#pragma unroll
      for (int c = 0; c < 16; ++c) { s1 += t[c]; s2 += t[c] * t[c]; }
      s1 += __shfl_xor(s1, 16);
      s2 += __shfl_xor(s2, 16);
      float m   = s1 * (1.0f / 32.0f);
      float var = s2 * (1.0f / 32.0f) - m * m;
      float inv = rsqrtf(var + 1e-5f);
#pragma unroll
      for (int q = 0; q < 4; ++q) {
        float4 o;
        o.x = (t[q * 4 + 0] - m) * inv * en_g[c0 + q * 4 + 0] + en_b[c0 + q * 4 + 0];
        o.y = (t[q * 4 + 1] - m) * inv * en_g[c0 + q * 4 + 1] + en_b[c0 + q * 4 + 1];
        o.z = (t[q * 4 + 2] - m) * inv * en_g[c0 + q * 4 + 2] + en_b[c0 + q * 4 + 2];
        o.w = (t[q * 4 + 3] - m) * inv * en_g[c0 + q * 4 + 3] + en_b[c0 + q * 4 + 3];
        ((float4*)(out_e + eoff))[q] = o;
      }
    }
  }

  // ---- fold per-wave softmax state to LDS ----------------------------------
  float accf[8];
#pragma unroll
  for (int nt = 0; nt < 8; ++nt) accf[nt] = acc8[nt] + __shfl_xor(acc8[nt], 16);
  if (lane < 16) {
#pragma unroll
    for (int nt = 0; nt < 8; ++nt) combacc[wave * 128 + nt * 16 + lane] = accf[nt];
  }
  if (lane == 0) {
#pragma unroll
    for (int hh = 0; hh < 4; ++hh) { combm[wave * 4 + hh] = mh[hh]; combl[wave * 4 + hh] = lh[hh]; }
  }
  float c0r = cw0, c1r = cw1;
#pragma unroll
  for (int off = 16; off >= 1; off >>= 1) { c0r += __shfl_xor(c0r, off); c1r += __shfl_xor(c1r, off); }
  if (lane == 0) { combcw[wave * 2] = c0r; combcw[wave * 2 + 1] = c1r; }
  __syncthreads();

  if (tid == 0) {
    for (int hh = 0; hh < 4; ++hh) {
      float m = combm[hh];
      for (int w = 1; w < 4; ++w) m = fmaxf(m, combm[w * 4 + hh]);
      float l = 0.f;
      for (int w = 0; w < 4; ++w) l += combl[w * 4 + hh] * __expf(combm[w * 4 + hh] - m);
      gm[hh] = m; gl[hh] = l;
    }
    float sx = 0.f, sy = 0.f;
    for (int w = 0; w < 4; ++w) { sx += combcw[w * 2]; sy += combcw[w * 2 + 1]; }
    float cs = coord_scale[0];
    out_x[((size_t)b * NN + i) * 2 + 0] = xi0 + cs * sx;
    out_x[((size_t)b * NN + i) * 2 + 1] = xi1 + cs * sy;
  }
  __syncthreads();

  // ---- messages_attended -> node update (tiny GEMV, plain VALU) ------------
  {
    int hh = tid >> 5;
    float s = 0.f;
    for (int w = 0; w < 4; ++w)
      s += combacc[w * 128 + tid] * __expf(combm[w * 4 + hh] - gm[hh]);
    hin[64 + tid] = s * fast_rcp(gl[hh]);
  }
  if (tid < 64) hin[tid] = hi_row[tid];
  __syncthreads();
  {
    float s = node_b1[tid];
    for (int k = 0; k < 192; ++k) s += hin[k] * node_W1[k * 128 + tid];
    tbuf[tid] = silu(s);
  }
  __syncthreads();
  if (tid == 0) {
    float m = 0.f; for (int k = 0; k < 128; ++k) m += tbuf[k]; m *= (1.0f / 128.0f);
    float v = 0.f; for (int k = 0; k < 128; ++k) { float d = tbuf[k] - m; v += d * d; }
    stats[0] = m; stats[1] = rsqrtf(v * (1.0f / 128.0f) + 1e-5f);
  }
  __syncthreads();
  tbuf[tid] = (tbuf[tid] - stats[0]) * stats[1] * node_ln_g[tid] + node_ln_b[tid];
  __syncthreads();
  if (tid < 64) {
    float s = node_b2[tid];
    for (int k = 0; k < 128; ++k) s += tbuf[k] * node_W2[k * 64 + tid];
    hres[tid] = s + hi_row[tid];
  }
  __syncthreads();
  if (tid == 0) {
    float m = 0.f; for (int k = 0; k < 64; ++k) m += hres[k]; m *= (1.0f / 64.0f);
    float v = 0.f; for (int k = 0; k < 64; ++k) { float d = hres[k] - m; v += d * d; }
    stats[2] = m; stats[3] = rsqrtf(v * (1.0f / 64.0f) + 1e-5f);
  }
  __syncthreads();
  if (tid < 64)
    out_h[((size_t)b * NN + i) * 64 + tid] =
        (hres[tid] - stats[2]) * stats[3] * nn_g[tid] + nn_b[tid];
}

// ---- host launcher ----------------------------------------------------------
extern "C" void kernel_launch(void* const* d_in, const int* in_sizes, int n_in,
                              void* d_out, int out_size, void* d_ws, size_t ws_size,
                              hipStream_t stream) {
  (void)in_sizes; (void)n_in; (void)out_size; (void)ws_size;
  const float* h        = (const float*)d_in[0];
  const float* x        = (const float*)d_in[1];
  const float* e        = (const float*)d_in[2];
  const float* msg_W1   = (const float*)d_in[3];
  const float* msg_b1   = (const float*)d_in[4];
  const float* msg_ln_g = (const float*)d_in[5];
  const float* msg_ln_b = (const float*)d_in[6];
  const float* msg_W2   = (const float*)d_in[7];
  const float* msg_b2   = (const float*)d_in[8];
  const float* msg_W3   = (const float*)d_in[9];
  const float* msg_b3   = (const float*)d_in[10];
  const float* attn_W1  = (const float*)d_in[11];
  const float* attn_b1  = (const float*)d_in[12];
  const float* attn_W2  = (const float*)d_in[13];
  const float* attn_b2  = (const float*)d_in[14];
  const float* val_W    = (const float*)d_in[15];
  const float* val_b    = (const float*)d_in[16];
  const float* coord_W1 = (const float*)d_in[17];
  const float* coord_b1 = (const float*)d_in[18];
  const float* coord_W2 = (const float*)d_in[19];
  const float* node_W1  = (const float*)d_in[20];
  const float* node_b1  = (const float*)d_in[21];
  const float* node_ln_g= (const float*)d_in[22];
  const float* node_ln_b= (const float*)d_in[23];
  const float* node_W2  = (const float*)d_in[24];
  const float* node_b2  = (const float*)d_in[25];
  const float* edge_W1  = (const float*)d_in[26];
  const float* edge_b1  = (const float*)d_in[27];
  const float* edge_ln_g= (const float*)d_in[28];
  const float* edge_ln_b= (const float*)d_in[29];
  const float* edge_W2  = (const float*)d_in[30];
  const float* edge_b2  = (const float*)d_in[31];
  const float* nn_g     = (const float*)d_in[32];
  const float* nn_b     = (const float*)d_in[33];
  const float* en_g     = (const float*)d_in[34];
  const float* en_b     = (const float*)d_in[35];
  const float* coord_sc = (const float*)d_in[36];

  _Float16* wp = (_Float16*)d_ws;

  struct Job { const float* W; int K; int Kpad; int N; size_t off; };
  const Job jobs[8] = {
      {msg_W1,   161, 192, 128, OFF_MSG_W1},
      {msg_W2,   128, 128, 128, OFF_MSG_W2},
      {msg_W3,   128, 128, 128, OFF_MSG_W3},
      {attn_W1,  128, 128,  64, OFF_ATTN_W1},
      {val_W,    128, 128, 128, OFF_VAL_W},
      {coord_W1, 128, 128,  64, OFF_COORD_W1},
      {edge_W1,  160, 160, 128, OFF_EDGE_W1},
      {edge_W2,  128, 128,  32, OFF_EDGE_W2},
  };
  for (int q = 0; q < 8; ++q) {
    int total = (jobs[q].Kpad >> 5) * (jobs[q].N >> 4) * 512;
    pack_weights<<<(total + 255) / 256, 256, 0, stream>>>(
        jobs[q].W, wp + jobs[q].off, jobs[q].K, jobs[q].Kpad, jobs[q].N);
  }

  float* out   = (float*)d_out;
  float* out_h = out;            // 2*512*64
  float* out_x = out + 65536;    // 2*512*2
  float* out_e = out + 67584;    // 2*512*512*32

  egnn_main<<<2 * NN, 128, SMEM_BYTES, stream>>>(
      h, x, e, wp,
      msg_b1, msg_ln_g, msg_ln_b, msg_b2, msg_b3,
      attn_b1, attn_W2, attn_b2, val_b,
      coord_b1, coord_W2,
      edge_b1, edge_ln_g, edge_ln_b, edge_b2, en_g, en_b,
      node_W1, node_b1, node_ln_g, node_ln_b, node_W2, node_b2,
      nn_g, nn_b, coord_sc,
      out_h, out_x, out_e);
}
